// ChunkAwareAttention_17489106829529
// MI455X (gfx1250) — matrix operand order repaired and checked by reading, not compiled, not run on hardware
//
#include <hip/hip_runtime.h>

// ---------------------------------------------------------------------------
// ChunkAwareAttention for MI455X (gfx1250, wave32, WMMA f16->f32, TDM, async)
// B=8 T=8192 D=512 H=8 DK=64 ; C=64 P=128 M=32 W=224 ; chunks n=128
// ---------------------------------------------------------------------------

typedef __attribute__((ext_vector_type(16))) _Float16 v16h;
typedef __attribute__((ext_vector_type(8)))  float    v8f;
typedef __attribute__((ext_vector_type(2)))  _Float16 h2;
typedef __attribute__((ext_vector_type(4)))  int          v4i;
typedef __attribute__((ext_vector_type(4)))  unsigned int u32x4;
typedef __attribute__((ext_vector_type(8)))  unsigned int u32x8;

union FH { v16h v; uint4 u[2]; _Float16 h[16]; };
union U4 { uint4 u; h2 p[4]; _Float16 h[8]; };

#if __has_builtin(__builtin_amdgcn_global_load_async_to_lds_b128) && \
    __has_builtin(__builtin_amdgcn_s_wait_asynccnt)
#define HAVE_ASYNC_LDS 1
#endif

#define AS1 __attribute__((address_space(1)))
#define AS3 __attribute__((address_space(3)))

__device__ __forceinline__ void wait_tensorcnt0() {
#if __has_builtin(__builtin_amdgcn_s_wait_tensorcnt)
  __builtin_amdgcn_s_wait_tensorcnt(0);
#else
  asm volatile("s_wait_tensorcnt 0x0" ::: "memory");
#endif
}

// Issue a 2D TDM tile load: global [rows x 64 halves, row stride 64] ->
// LDS rows padded to 72 halves (pad 4 dwords every 32 dwords).
__device__ __forceinline__ void tdm_load_2d(const _Float16* gsrc, void* lds,
                                            int rows) {
  unsigned long long ga = (unsigned long long)gsrc;
  unsigned lo = (unsigned)(((unsigned long long)(size_t)lds) & 0xffffffffull);
  u32x4 g0;
  g0[0] = 1u;                                   // count=1, user mode
  g0[1] = lo;                                   // lds_addr (bytes)
  g0[2] = (unsigned)(ga & 0xffffffffull);       // global_addr[31:0]
  g0[3] = (unsigned)((ga >> 32) & 0x01ffffffull) | (2u << 30);  // [56:32]|type=2
  u32x8 g1;
  g1[0] = (1u << 16)    // data_size = 2 bytes
        | (1u << 20)    // pad_enable
        | (4u << 22)    // pad_interval: 32 dwords (= 64 halves)
        | (3u << 25);   // pad_amount: 4 dwords   (= 8 halves -> stride 72)
  g1[1] = (64u << 16);              // tensor_dim0[15:0]=64 in bits 63:48
  g1[2] = 0u;                       // tensor_dim0[31:16]=0, tensor_dim1[15:0]=0
  g1[3] = 0x10u | (64u << 16);      // tensor_dim1[31:16]=0x10 (big), tile_dim0=64
  g1[4] = (unsigned)rows;           // tile_dim1 = rows, tile_dim2 = 0
  g1[5] = 64u;                      // tensor_dim0_stride[31:0] = 64
  g1[6] = 0u;
  g1[7] = 0u;
  asm volatile("tensor_load_to_lds %0, %1" ::"s"(g0), "s"(g1) : "memory");
}

__device__ __forceinline__ v8f wmma_f16(v16h a, v16h b, v8f c) {
  return __builtin_amdgcn_wmma_f32_16x16x32_f16(false, a, false, b, (short)0, c,
                                                false, false);
}

// A fragment (16x32 f16) from row-major [m][k]; ld multiple of 8 halves,
// s 16B-aligned.  CDNA5: lane l: M=l&15; K groups split by l>>4.
__device__ __forceinline__ v16h frag_a_h(const _Float16* s, int ld) {
  const int l = threadIdx.x & 31;
  const _Float16* p = s + (l & 15) * ld + 8 * (l >> 4);
  FH f;
  f.u[0] = *(const uint4*)p;
  f.u[1] = *(const uint4*)(p + 16);
  return f.v;
}

// A fragment from an f32 array (probs), converted on the fly.
__device__ __forceinline__ v16h frag_a_f(const float* s, int ld) {
  const int l = threadIdx.x & 31;
  const float* p = s + (l & 15) * ld + 8 * (l >> 4);
  float4 x0 = *(const float4*)p;
  float4 x1 = *(const float4*)(p + 4);
  float4 y0 = *(const float4*)(p + 16);
  float4 y1 = *(const float4*)(p + 20);
  v16h a;
  a[0] = (_Float16)x0.x; a[1] = (_Float16)x0.y; a[2] = (_Float16)x0.z; a[3] = (_Float16)x0.w;
  a[4] = (_Float16)x1.x; a[5] = (_Float16)x1.y; a[6] = (_Float16)x1.z; a[7] = (_Float16)x1.w;
  a[8] = (_Float16)y0.x; a[9] = (_Float16)y0.y; a[10] = (_Float16)y0.z; a[11] = (_Float16)y0.w;
  a[12] = (_Float16)y1.x; a[13] = (_Float16)y1.y; a[14] = (_Float16)y1.z; a[15] = (_Float16)y1.w;
  return a;
}

// B fragment (32x16) where the source holds B transposed: B[k][n] = s[n*ld+k].
// Two 8-half groups per lane (k=8h.., 16+8h..) -> two b128 loads.  ld multiple of 8 halves.
__device__ __forceinline__ v16h frag_bt_h(const _Float16* s, int ld) {
  const int l = threadIdx.x & 31;
  const _Float16* p = s + (l & 15) * ld + 8 * (l >> 4);
  FH f;
  f.u[0] = *(const uint4*)p;
  f.u[1] = *(const uint4*)(p + 16);
  return f.v;
}

// ---------------------------------------------------------------------------
__global__ __launch_bounds__(256) void cvt_f16_kernel(const float* __restrict__ src,
                                                      _Float16* __restrict__ dst,
                                                      int n) {
  int i = blockIdx.x * 256 + threadIdx.x;
  if (i < n) dst[i] = (_Float16)src[i];
}

__global__ __launch_bounds__(256) void pack_qkv_kernel(
    const float* __restrict__ Wq, const float* __restrict__ Wk,
    const float* __restrict__ Wv, const float* __restrict__ bq,
    const float* __restrict__ bk, const float* __restrict__ bv,
    _Float16* __restrict__ Wqkv, float* __restrict__ bqkv) {
  int i = blockIdx.x * 256 + threadIdx.x;
  if (i < 512 * 512) {
    int kk = i >> 9, c = i & 511;
    size_t ro = (size_t)kk * 1536;
    Wqkv[ro + c]        = (_Float16)Wq[i];
    Wqkv[ro + 512 + c]  = (_Float16)Wk[i];
    Wqkv[ro + 1024 + c] = (_Float16)Wv[i];
  }
  if (i < 512) {
    bqkv[i]        = bq[i];
    bqkv[512 + i]  = bk[i];
    bqkv[1024 + i] = bv[i];
  }
}

// ---------------------------------------------------------------------------
__global__ __launch_bounds__(256) void ln_kernel(const float* __restrict__ x,
                                                 const float* __restrict__ g,
                                                 const float* __restrict__ bta,
                                                 _Float16* __restrict__ xn) {
  const int row = blockIdx.x;
  const int tid = threadIdx.x;
  const float* xr = x + (size_t)row * 512;
  float a = xr[tid], b = xr[tid + 256];
  float s = a + b, ss = a * a + b * b;
#pragma unroll
  for (int o = 16; o > 0; o >>= 1) {
    s += __shfl_down(s, o);
    ss += __shfl_down(ss, o);
  }
  __shared__ float rs[8], rq[8];
  __shared__ float mu_s, ri_s;
  if ((tid & 31) == 0) { rs[tid >> 5] = s; rq[tid >> 5] = ss; }
  __syncthreads();
  if (tid == 0) {
    float S = 0.f, Q = 0.f;
#pragma unroll
    for (int i = 0; i < 8; ++i) { S += rs[i]; Q += rq[i]; }
    float mu = S * (1.f / 512.f);
    float var = Q * (1.f / 512.f) - mu * mu;
    mu_s = mu;
    ri_s = rsqrtf(var + 1e-5f);
  }
  __syncthreads();
  float mu = mu_s, ri = ri_s;
  xn[(size_t)row * 512 + tid]       = (_Float16)((a - mu) * ri * g[tid] + bta[tid]);
  xn[(size_t)row * 512 + tid + 256] = (_Float16)((b - mu) * ri * g[tid + 256] + bta[tid + 256]);
}

// ---------------------------------------------------------------------------
// WMMA GEMM  C[M,N] = A[M,K]@B[K,N] + bias ; block tile 128x128, 8 waves
// (wave tile 32x64, acc 2x4), K-step 64 -> 16 WMMAs between barriers.
// A staged via async global->LDS; B staged transposed (contiguous b128 frags).
//   mode 0: f32 row-major ; mode 1: head-split f16 (q|k|v sections)
// ---------------------------------------------------------------------------
__global__ __launch_bounds__(256) void gemm_wmma_kernel(
    const _Float16* __restrict__ A, const _Float16* __restrict__ Bm,
    const float* __restrict__ bias, float* __restrict__ Cf,
    _Float16* __restrict__ O0, _Float16* __restrict__ O1,
    _Float16* __restrict__ O2, int M, int N, int K, int mode) {
  __shared__ _Float16 As[128][72];    // 128x64 tile, stride 72 (144B rows)
  __shared__ _Float16 Bst[128][72];   // B tile transposed [n][k], 128x64
  const int tid = threadIdx.x;
  const int wid = tid >> 5;
  const int wm = wid & 3, wn = wid >> 2;
  const int rowBase = blockIdx.y * 128;
  const int colBase = blockIdx.x * 128;

  const v8f vzero = {0.f, 0.f, 0.f, 0.f, 0.f, 0.f, 0.f, 0.f};
  v8f acc[2][4];
#pragma unroll
  for (int mi = 0; mi < 2; ++mi)
#pragma unroll
    for (int ni = 0; ni < 4; ++ni) acc[mi][ni] = vzero;

  for (int k0 = 0; k0 < K; k0 += 64) {
    // ---- stage A tile (128x64 halves = 1024 x 16B chunks) ----
#pragma unroll
    for (int i = 0; i < 4; ++i) {
      int idx = tid + i * 256;
      int r = idx >> 3, c = (idx & 7) << 3;
      const _Float16* src = A + (size_t)(rowBase + r) * K + k0 + c;
#ifdef HAVE_ASYNC_LDS
      __builtin_amdgcn_global_load_async_to_lds_b128(
          (AS1 v4i*)src, (AS3 v4i*)&As[r][c], 0, 0);
#else
      *(uint4*)&As[r][c] = *(const uint4*)src;
#endif
    }
    // ---- stage B tile transposed: Bst[n][k], 64(k) x 128(n) ----
#pragma unroll
    for (int i = 0; i < 4; ++i) {
      int idx = tid + i * 256;
      int r = idx >> 4, c = (idx & 15) << 3;
      U4 t;
      t.u = *(const uint4*)(Bm + (size_t)(k0 + r) * N + colBase + c);
#pragma unroll
      for (int e = 0; e < 8; ++e) Bst[c + e][r] = t.h[e];
    }
#ifdef HAVE_ASYNC_LDS
    __builtin_amdgcn_s_wait_asynccnt(0);
#endif
    __syncthreads();

#pragma unroll
    for (int kk = 0; kk < 64; kk += 32) {
      v16h af0 = frag_a_h(&As[wm * 32][kk], 72);
      v16h af1 = frag_a_h(&As[wm * 32 + 16][kk], 72);
      v16h bf[4];
#pragma unroll
      for (int ni = 0; ni < 4; ++ni)
        bf[ni] = frag_bt_h(&Bst[wn * 64 + ni * 16][kk], 72);
#pragma unroll
      for (int ni = 0; ni < 4; ++ni) {
        acc[0][ni] = wmma_f16(af0, bf[ni], acc[0][ni]);
        acc[1][ni] = wmma_f16(af1, bf[ni], acc[1][ni]);
      }
    }
    __syncthreads();
  }

  const int l = tid & 31, hi = l >> 4, nlo = l & 15;
#pragma unroll
  for (int mi = 0; mi < 2; ++mi) {
#pragma unroll
    for (int ni = 0; ni < 4; ++ni) {
      int row0 = rowBase + wm * 32 + mi * 16;
      int col = colBase + wn * 64 + ni * 16 + nlo;
      float bv = bias ? bias[col] : 0.f;
      if (mode == 0) {
        float* dst = Cf + (size_t)(row0 + 8 * hi) * N + col;
#pragma unroll
        for (int v = 0; v < 8; ++v) dst[(size_t)v * N] = acc[mi][ni][v] + bv;
      } else {
        // row -> (b = row>>13, t = row&8191); b constant within a 16-row tile
        int bb = row0 >> 13, t0 = row0 & 8191;
        int sct = col >> 9, h = (col >> 6) & 7, dk = col & 63;
        _Float16* Ob = (sct == 0) ? O0 : ((sct == 1) ? O1 : O2);
        _Float16* dst =
            Ob + (((size_t)(bb * 8 + h)) * 8192 + t0 + 8 * hi) * 64 + dk;
#pragma unroll
        for (int v = 0; v < 8; ++v)
          dst[(size_t)v * 64] = (_Float16)(acc[mi][ni][v] + bv);
      }
    }
  }
}

// ---------------------------------------------------------------------------
// Chunked attention: block per (chunk, h, b); 256 threads = 8 waves.
// Q arrives via the Tensor Data Mover (padded to stride 72); Ksum = K + PK
// staged with packed-f16 adds; V staged transposed so P@V B-fragments are
// contiguous.  scores = Q@Ksum^T * 0.125, boundary mask, softmax, P@V.
// ---------------------------------------------------------------------------
__global__ __launch_bounds__(256) void attn_kernel(
    const _Float16* __restrict__ q, const _Float16* __restrict__ k,
    const _Float16* __restrict__ v, const _Float16* __restrict__ pk,
    _Float16* __restrict__ aout) {
  const int ch = blockIdx.x, h = blockIdx.y, b = blockIdx.z;
  const int qs = ch * 64;
  const int bh = b * 8 + h;
  const int tid = threadIdx.x;
  const int wid = tid >> 5;
  const int l = tid & 31, hi = l >> 4, nlo = l & 15;

  __shared__ _Float16 Qs[64][72];    //  9.2 KB (TDM, pad -> stride 72)
  __shared__ _Float16 Ks[224][72];   // 32.3 KB (K + PK)
  __shared__ _Float16 Vst[64][232];  // 29.7 KB (V transposed: [dk][w])
  __shared__ float    Sc[64][228];   // 58.4 KB scores -> probs (f32)

  // ---- TDM: Q chunk (64 rows); wave 0 issues ----
  if (wid == 0) {
    const _Float16* qg = q + ((size_t)bh * 8192 + qs) * 64;
    tdm_load_2d(qg, &Qs[0][0], 64);
    wait_tensorcnt0();
  }
  // ---- stage Ksum = K + PK ; V transposed ; window [qs-128, qs+96) ----
  for (int i = tid; i < 224 * 8; i += 256) {
    int w = i >> 3, c = (i & 7) << 3;
    int kv = qs - 128 + w;  // may be OOB; rows are masked in softmax
    U4 a, p, vv;
    a.u = *(const uint4*)(k + ((size_t)bh * 8192 + kv) * 64 + c);
    p.u = *(const uint4*)(pk + ((size_t)h * 8192 + kv) * 64 + c);
    vv.u = *(const uint4*)(v + ((size_t)bh * 8192 + kv) * 64 + c);
#pragma unroll
    for (int j = 0; j < 4; ++j) a.p[j] += p.p[j];
    *(uint4*)&Ks[w][c] = a.u;
#pragma unroll
    for (int e = 0; e < 8; ++e) Vst[c + e][w] = vv.h[e];
  }
  __syncthreads();

  // ---- scores: 64x224 = 4x14 tiles; wave -> (row tile, 7 col tiles) ----
  const v8f vzero = {0.f, 0.f, 0.f, 0.f, 0.f, 0.f, 0.f, 0.f};
  {
    const int rm = wid >> 1;
    v16h af0 = frag_a_h(&Qs[rm * 16][0], 72);   // hoisted: reused for all j
    v16h af1 = frag_a_h(&Qs[rm * 16][32], 72);
    for (int j = 0; j < 7; ++j) {
      int cn = (wid & 1) * 7 + j;
      v8f acc = vzero;
      acc = wmma_f16(af0, frag_bt_h(&Ks[cn * 16][0], 72), acc);
      acc = wmma_f16(af1, frag_bt_h(&Ks[cn * 16][32], 72), acc);
      int col = cn * 16 + nlo;
      int kv = qs - 128 + col;
      bool ok = (kv >= 0) && (kv < 8192);
#pragma unroll
      for (int vv = 0; vv < 8; ++vv)
        Sc[rm * 16 + vv + 8 * hi][col] = ok ? acc[vv] * 0.125f : -1e30f;
    }
  }
  __syncthreads();

  // ---- softmax: 4 lanes per row, shfl_xor reduce ----
  {
    int row = tid >> 2, seg = tid & 3;
    float mx = -1e30f;
    for (int w = seg; w < 224; w += 4) mx = fmaxf(mx, Sc[row][w]);
    mx = fmaxf(mx, __shfl_xor(mx, 1));
    mx = fmaxf(mx, __shfl_xor(mx, 2));
    float sum = 0.f;
    for (int w = seg; w < 224; w += 4) {
      float e = __expf(Sc[row][w] - mx);
      Sc[row][w] = e;
      sum += e;
    }
    sum += __shfl_xor(sum, 1);
    sum += __shfl_xor(sum, 2);
    float inv = 1.f / sum;
    for (int w = seg; w < 224; w += 4) Sc[row][w] *= inv;
  }
  __syncthreads();

  // ---- out = P(64x224) @ V(224x64): 4x4 tiles; wave -> 2 tiles ----
  {
    const int tm = wid >> 1;
    v16h paf[7];                       // hoisted: reused for both jj tiles
#pragma unroll
    for (int kk = 0; kk < 7; ++kk) paf[kk] = frag_a_f(&Sc[tm * 16][kk * 32], 228);
#pragma unroll
    for (int jj = 0; jj < 2; ++jj) {
      int tn = (wid & 1) * 2 + jj;
      v8f acc = vzero;
#pragma unroll
      for (int kk = 0; kk < 7; ++kk)
        acc = wmma_f16(paf[kk], frag_bt_h(&Vst[tn * 16][kk * 32], 232), acc);
      _Float16* og = aout + ((size_t)b * 8192 + qs + tm * 16 + 8 * hi) * 512 +
                     h * 64 + tn * 16 + nlo;
#pragma unroll
      for (int vv = 0; vv < 8; ++vv)
        og[(size_t)vv * 512] = (_Float16)acc[vv];
    }
  }
}

// ---------------------------------------------------------------------------
extern "C" void kernel_launch(void* const* d_in, const int* in_sizes, int n_in,
                              void* d_out, int out_size, void* d_ws,
                              size_t ws_size, hipStream_t stream) {
  const float* x    = (const float*)d_in[0];
  const float* pos  = (const float*)d_in[1];
  // d_in[2] = mask : analytically redundant (window == chunk mask), unused
  const float* ln_g = (const float*)d_in[3];
  const float* ln_b = (const float*)d_in[4];
  const float* Wq   = (const float*)d_in[5];
  const float* bq   = (const float*)d_in[6];
  const float* Wk   = (const float*)d_in[7];
  const float* bk   = (const float*)d_in[8];
  const float* Wv   = (const float*)d_in[9];
  const float* bv   = (const float*)d_in[10];
  const float* Wpos = (const float*)d_in[11];
  const float* Wout = (const float*)d_in[12];
  const float* bout = (const float*)d_in[13];
  float* out = (float*)d_out;

  char* ws = (char*)d_ws;
  _Float16* xn   = (_Float16*)(ws + 0ull);          // 64 MB
  _Float16* qb   = (_Float16*)(ws + 67108864ull);   // 64 MB [B*H][T][64]
  _Float16* kb   = (_Float16*)(ws + 134217728ull);  // 64 MB
  _Float16* vb   = (_Float16*)(ws + 201326592ull);  // 64 MB
  _Float16* ab   = (_Float16*)(ws + 268435456ull);  // 64 MB attention out
  _Float16* xpos = (_Float16*)(ws + 335544320ull);  //  8 MB
  _Float16* pkb  = (_Float16*)(ws + 343932928ull);  //  8 MB [H][T][64]
  _Float16* wqkv = (_Float16*)(ws + 352321536ull);  // 1.5 MB [512][1536]
  _Float16* wpos = (_Float16*)(ws + 353894400ull);  // 0.5 MB
  _Float16* wout = (_Float16*)(ws + 354418688ull);  // 0.5 MB
  float*    bqkv = (float*)   (ws + 354942976ull);  // 6 KB

  cvt_f16_kernel<<<(8192 * 512 + 255) / 256, 256, 0, stream>>>(pos, xpos,
                                                               8192 * 512);
  cvt_f16_kernel<<<(512 * 512 + 255) / 256, 256, 0, stream>>>(Wpos, wpos,
                                                              512 * 512);
  cvt_f16_kernel<<<(512 * 512 + 255) / 256, 256, 0, stream>>>(Wout, wout,
                                                              512 * 512);
  pack_qkv_kernel<<<(512 * 512 + 255) / 256, 256, 0, stream>>>(
      Wq, Wk, Wv, bq, bk, bv, wqkv, bqkv);

  ln_kernel<<<65536, 256, 0, stream>>>(x, ln_g, ln_b, xn);

  // QKV: [65536,512] @ [512,1536]  (block tile 128x128)
  gemm_wmma_kernel<<<dim3(1536 / 128, 65536 / 128), 256, 0, stream>>>(
      xn, wqkv, bqkv, nullptr, qb, kb, vb, 65536, 1536, 512, 1);

  // positional keys: [8192,512] @ [512,512]
  gemm_wmma_kernel<<<dim3(512 / 128, 8192 / 128), 256, 0, stream>>>(
      xpos, wpos, nullptr, nullptr, pkb, pkb, pkb, 8192, 512, 512, 1);

  attn_kernel<<<dim3(128, 8, 8), 256, 0, stream>>>(qb, kb, vb, pkb, ab);

  // output projection: [65536,512] @ [512,512] + bout -> f32
  gemm_wmma_kernel<<<dim3(512 / 128, 65536 / 128), 256, 0, stream>>>(
      ab, wout, bout, out, nullptr, nullptr, nullptr, 65536, 512, 512, 0);
}
